// GNNLLMFusion_3023656976579
// MI455X (gfx1250) — matrix operand order repaired and checked by reading, not compiled, not run on hardware
//
#include <hip/hip_runtime.h>
#include <hip/hip_bf16.h>
#include <stdint.h>

// ---------------- problem constants ----------------
#define NN      50000       // nodes
#define NE      500000      // edges
#define NT      550000      // edges + self loops
#define INDIM   896
#define SDIM    128         // STRUCT
#define SEMDIM  768
#define HID     128
#define NHEADS  4
#define F1      512         // NHEADS*HID

typedef __attribute__((ext_vector_type(16))) _Float16 v16h;
typedef __attribute__((ext_vector_type(8)))  float    v8f;

struct Frag32B { uint4 lo, hi; };

// A-matrix 16x32 f16 fragment (ISA 7.12.2): lanes 0-15 -> M=lane, K chunks
// {hl*8..hl*8+7} and {16+hl*8..23+hl*8} where hl = lane>>4.
__device__ __forceinline__ v16h load_a_frag(const _Float16* rowk, int hl) {
  Frag32B t;
  t.lo = *(const uint4*)(rowk + hl * 8);
  t.hi = *(const uint4*)(rowk + 16 + hl * 8);
  return __builtin_bit_cast(v16h, t);
}
// B-matrix 32x16 f16 fragment from pre-transposed Bt (N x K row-major):
// lane holds column n = lane&15, K chunks {hl*8..hl*8+7} and {16+hl*8..23+hl*8}.
__device__ __forceinline__ v16h load_b_frag(const _Float16* rowk, int hl) {
  Frag32B t;
  t.lo = *(const uint4*)(rowk + hl * 8);
  t.hi = *(const uint4*)(rowk + 16 + hl * 8);
  return __builtin_bit_cast(v16h, t);
}

// ---------------- WMMA GEMM: C(MxNc,f32) = A(MxK,f16) * Bt(NcxK,f16)^T ----
__global__ __launch_bounds__(256)
void gemm_wmma_f16(const _Float16* __restrict__ A, int lda,
                   const _Float16* __restrict__ Bt, int ldb,
                   float* __restrict__ C, int ldc,
                   int M, int K, int Ntiles) {
  const int lane = threadIdx.x & 31;
  const int wid  = blockIdx.x * (blockDim.x >> 5) + (threadIdx.x >> 5);
  const int Mtiles = (M + 31) >> 5;
  if (wid >= Mtiles * Ntiles) return;           // wave-uniform exit, EXEC stays full
  const int mt = wid / Ntiles, nt = wid % Ntiles;
  const int r = lane & 15, hl = lane >> 4;

  int m0 = mt * 32 + r;
  int m1 = m0 + 16;
  int mA0 = m0 < M ? m0 : M - 1;                // clamp loads, mask stores
  int mA1 = m1 < M ? m1 : M - 1;
  const _Float16* a0p = A + (size_t)mA0 * lda;
  const _Float16* a1p = A + (size_t)mA1 * lda;
  const _Float16* b0p = Bt + (size_t)(nt * 32 + r) * ldb;
  const _Float16* b1p = b0p + (size_t)16 * ldb;

  v8f c00 = {}, c01 = {}, c10 = {}, c11 = {};
  for (int kk = 0; kk < K; kk += 32) {
    v16h a0 = load_a_frag(a0p + kk, hl);
    v16h a1 = load_a_frag(a1p + kk, hl);
    v16h b0 = load_b_frag(b0p + kk, hl);
    v16h b1 = load_b_frag(b1p + kk, hl);
    c00 = __builtin_amdgcn_wmma_f32_16x16x32_f16(false, a0, false, b0, (short)0, c00, false, false);
    c01 = __builtin_amdgcn_wmma_f32_16x16x32_f16(false, a0, false, b1, (short)0, c01, false, false);
    c10 = __builtin_amdgcn_wmma_f32_16x16x32_f16(false, a1, false, b0, (short)0, c10, false, false);
    c11 = __builtin_amdgcn_wmma_f32_16x16x32_f16(false, a1, false, b1, (short)0, c11, false, false);
  }
  // C/D layout: VGPR v -> row M = v + 8*hl, col N = lane&15 (ISA 7.12.2)
  const int n0 = nt * 32 + r, n1 = n0 + 16;
#pragma unroll
  for (int v = 0; v < 8; ++v) {
    int row0 = mt * 32 + v + 8 * hl;
    int row1 = row0 + 16;
    if (row0 < M) {
      C[(size_t)row0 * ldc + n0] = c00[v];
      C[(size_t)row0 * ldc + n1] = c01[v];
    }
    if (row1 < M) {
      C[(size_t)row1 * ldc + n0] = c10[v];
      C[(size_t)row1 * ldc + n1] = c11[v];
    }
  }
}

// ---------------- helpers ----------------
__device__ __forceinline__ unsigned fkey(float f) {           // order-preserving
  unsigned u = __float_as_uint(f);
  return (u & 0x80000000u) ? ~u : (u | 0x80000000u);
}
__device__ __forceinline__ float fdecode(unsigned k) {
  unsigned u = (k & 0x80000000u) ? (k & 0x7FFFFFFFu) : ~k;
  return __uint_as_float(u);
}
__device__ __forceinline__ float sigmoidf_(float x) { return 1.0f / (1.0f + expf(-x)); }
__device__ __forceinline__ float eluf_(float x) { return x > 0.0f ? x : (expf(x) - 1.0f); }

// ---------------- small kernels ----------------
__global__ void k_zero(float* p, long n) {
  long i = (long)blockIdx.x * blockDim.x + threadIdx.x;
  long stride = (long)gridDim.x * blockDim.x;
  for (; i < n; i += stride) p[i] = 0.0f;
}

__global__ void k_f32_to_f16(const float* __restrict__ src, _Float16* __restrict__ dst, long n) {
  long i = (long)blockIdx.x * blockDim.x + threadIdx.x;
  long stride = (long)gridDim.x * blockDim.x;
  for (; i < n; i += stride) dst[i] = (_Float16)src[i];
}

// transpose-convert: src (K x Nc f32) -> dst (Nc x K f16)
__global__ void k_transp(const float* __restrict__ src, _Float16* __restrict__ dst, int K, int Nc) {
  long n = (long)K * Nc;
  long i = (long)blockIdx.x * blockDim.x + threadIdx.x;
  long stride = (long)gridDim.x * blockDim.x;
  for (; i < n; i += stride) {
    int k = (int)(i / Nc), c = (int)(i % Nc);
    dst[(size_t)c * K + k] = (_Float16)src[i];
  }
}

// cross-attention fuse: wave per node; cat0[i] = [x_struct(f16) | sigmoid(Q.K)*V]
__global__ __launch_bounds__(256)
void k_fuse1(const float* __restrict__ x, const float* __restrict__ Qf,
             const float* __restrict__ Kf, const float* __restrict__ Vf,
             _Float16* __restrict__ cat0) {
  int wave = threadIdx.x >> 5, lane = threadIdx.x & 31;
  int i = blockIdx.x * 8 + wave;
  if (i >= NN) return;
  float s = 0.0f;
#pragma unroll
  for (int k = 0; k < 4; ++k) {
    int c = lane + k * 32;
    s += Qf[(size_t)i * HID + c] * Kf[(size_t)i * HID + c];
  }
#pragma unroll
  for (int m = 16; m >= 1; m >>= 1) s += __shfl_xor(s, m, 32);
  float sg = sigmoidf_(s);
#pragma unroll
  for (int k = 0; k < 4; ++k) {
    int c = lane + k * 32;
    cat0[(size_t)i * 256 + c]       = (_Float16)x[(size_t)i * INDIM + c];
    cat0[(size_t)i * 256 + 128 + c] = (_Float16)(sg * Vf[(size_t)i * HID + c]);
  }
}

// h = [h0 + bo | x_sem] in f16 (N x 896)
__global__ void k_build_h(const float* __restrict__ h0f, const float* __restrict__ bo,
                          const _Float16* __restrict__ x_h, _Float16* __restrict__ h_h) {
  long n = (long)NN * INDIM;
  long idx = (long)blockIdx.x * blockDim.x + threadIdx.x;
  long stride = (long)gridDim.x * blockDim.x;
  for (; idx < n; idx += stride) {
    long i = idx / INDIM; int j = (int)(idx % INDIM);
    h_h[idx] = (j < SDIM) ? (_Float16)(h0f[i * SDIM + j] + bo[j]) : x_h[idx];
  }
}

// per-node attention coefficients: thread per (node, head)
__global__ void k_acoef(const float* __restrict__ hw, const float* __restrict__ att_s,
                        const float* __restrict__ att_d, float* __restrict__ asrc,
                        float* __restrict__ adst, int heads) {
  long t = (long)blockIdx.x * blockDim.x + threadIdx.x;
  if (t >= (long)NN * heads) return;
  int i = (int)(t / heads), h = (int)(t % heads);
  const float* row = hw + (size_t)i * heads * HID + h * HID;
  float s1 = 0.0f, s2 = 0.0f;
  for (int c = 0; c < HID; ++c) {
    float v = row[c];
    s1 += v * att_s[h * HID + c];
    s2 += v * att_d[h * HID + c];
  }
  asrc[t] = s1; adst[t] = s2;
}

__device__ __forceinline__ void edge_ends(const int* ei, int e, int& s, int& d) {
  if (e < NE) { s = ei[e]; d = ei[NE + e]; } else { s = e - NE; d = e - NE; }
}

// alpha = leaky_relu(asrc[src]+adst[dst]); segment-max via uint-key atomicMax
__global__ void k_edge_alpha(const int* __restrict__ ei, const float* __restrict__ asrc,
                             const float* __restrict__ adst, float* __restrict__ alpha,
                             unsigned* __restrict__ amax, int heads) {
  int e = blockIdx.x * blockDim.x + threadIdx.x;
  if (e >= NT) return;
  int s, d; edge_ends(ei, e, s, d);
  for (int h = 0; h < heads; ++h) {
    float al = asrc[(size_t)s * heads + h] + adst[(size_t)d * heads + h];
    al = al > 0.0f ? al : 0.2f * al;
    alpha[(size_t)e * heads + h] = al;
    atomicMax(&amax[(size_t)d * heads + h], fkey(al));
  }
}

// ex = exp(alpha - amax[dst]); denom[dst] += ex (alpha overwritten with ex)
__global__ void k_edge_exp(const int* __restrict__ ei, float* __restrict__ alpha,
                           const unsigned* __restrict__ amax, float* __restrict__ denom,
                           int heads) {
  int e = blockIdx.x * blockDim.x + threadIdx.x;
  if (e >= NT) return;
  int s, d; edge_ends(ei, e, s, d);
  for (int h = 0; h < heads; ++h) {
    float am = fdecode(amax[(size_t)d * heads + h]);
    float ex = expf(alpha[(size_t)e * heads + h] - am);
    alpha[(size_t)e * heads + h] = ex;
    atomicAdd(&denom[(size_t)d * heads + h], ex);
  }
}

// out[dst] += hw[src] * att ; wave per edge
__global__ __launch_bounds__(256)
void k_edge_scatter(const int* __restrict__ ei, const float* __restrict__ hw,
                    const float* __restrict__ ex, const float* __restrict__ denom,
                    float* __restrict__ out, int heads) {
  int wave = threadIdx.x >> 5, lane = threadIdx.x & 31;
  int e = blockIdx.x * 8 + wave;
  if (e >= NT) return;
  int s, d; edge_ends(ei, e, s, d);
  float attv[NHEADS];
  for (int h = 0; h < heads; ++h)
    attv[h] = ex[(size_t)e * heads + h] / (denom[(size_t)d * heads + h] + 1e-16f);
  int F = heads * HID;
  for (int j = lane; j < F; j += 32)
    atomicAdd(&out[(size_t)d * F + j], hw[(size_t)s * F + j] * attv[j >> 7]);
}

// column sums of (A + bias): grid (ceil(C/256), CHUNKS)
__global__ void k_colsum(const float* __restrict__ A, const float* __restrict__ bias,
                         float* __restrict__ colsum, int C) {
  int j = blockIdx.x * blockDim.x + threadIdx.x;
  if (j >= C) return;
  int chunk = (NN + gridDim.y - 1) / gridDim.y;
  int r0 = blockIdx.y * chunk;
  int r1 = r0 + chunk < NN ? r0 + chunk : NN;
  float b = bias[j], acc = 0.0f;
  for (int r = r0; r < r1; ++r) acc += A[(size_t)r * C + j] + b;
  atomicAdd(&colsum[j], acc);
}

__global__ void k_colvar(const float* __restrict__ A, const float* __restrict__ bias,
                         const float* __restrict__ colsum, const float* __restrict__ ms,
                         float* __restrict__ varsum, int C) {
  int j = blockIdx.x * blockDim.x + threadIdx.x;
  if (j >= C) return;
  int chunk = (NN + gridDim.y - 1) / gridDim.y;
  int r0 = blockIdx.y * chunk;
  int r1 = r0 + chunk < NN ? r0 + chunk : NN;
  float b = bias[j], m = colsum[j] * (1.0f / NN) * ms[j], acc = 0.0f;
  for (int r = r0; r < r1; ++r) {
    float c = A[(size_t)r * C + j] + b - m;
    acc += c * c;
  }
  atomicAdd(&varsum[j], acc);
}

// normalize + elu; output f16 (next-GEMM input) or f32
__global__ void k_norm(const float* __restrict__ A, const float* __restrict__ bias,
                       const float* __restrict__ colsum, const float* __restrict__ varsum,
                       const float* __restrict__ ms, const float* __restrict__ w,
                       const float* __restrict__ b, void* out, int C, int f16out) {
  long n = (long)NN * C;
  long idx = (long)blockIdx.x * blockDim.x + threadIdx.x;
  long stride = (long)gridDim.x * blockDim.x;
  for (; idx < n; idx += stride) {
    int j = (int)(idx % C);
    float mean = colsum[j] * (1.0f / NN);
    float var  = varsum[j] * (1.0f / NN);
    float c = A[idx] + bias[j] - mean * ms[j];
    float o = eluf_(c * rsqrtf(var + 1e-5f) * w[j] + b[j]);
    if (f16out) ((_Float16*)out)[idx] = (_Float16)o;
    else        ((float*)out)[idx] = o;
  }
}

// final tiny GEMM: out (N x 2) = h3 (N x 128) @ Wn + bn
__global__ void k_final(const float* __restrict__ h3, const float* __restrict__ Wn,
                        const float* __restrict__ bn, float* __restrict__ out) {
  int i = blockIdx.x * blockDim.x + threadIdx.x;
  if (i >= NN) return;
  float a0 = bn[0], a1 = bn[1];
  const float* row = h3 + (size_t)i * HID;
  for (int j = 0; j < HID; ++j) {
    float v = row[j];
    a0 += v * Wn[j * 2 + 0];
    a1 += v * Wn[j * 2 + 1];
  }
  out[(size_t)i * 2 + 0] = a0;
  out[(size_t)i * 2 + 1] = a1;
}

// ---------------- host ----------------
extern "C" void kernel_launch(void* const* d_in, const int* in_sizes, int n_in,
                              void* d_out, int out_size, void* d_ws, size_t ws_size,
                              hipStream_t stream) {
  const float* x    = (const float*)d_in[0];
  const int*   ei   = (const int*)d_in[1];
  const float* Wq   = (const float*)d_in[2];
  const float* Wk   = (const float*)d_in[3];
  const float* Wv   = (const float*)d_in[4];
  const float* Wo   = (const float*)d_in[5];
  const float* bo   = (const float*)d_in[6];
  const float* W1   = (const float*)d_in[7];
  const float* as1  = (const float*)d_in[8];
  const float* ad1  = (const float*)d_in[9];
  const float* b1   = (const float*)d_in[10];
  const float* g1w  = (const float*)d_in[11];
  const float* g1b  = (const float*)d_in[12];
  const float* g1ms = (const float*)d_in[13];
  const float* W2   = (const float*)d_in[14];
  const float* as2  = (const float*)d_in[15];
  const float* ad2  = (const float*)d_in[16];
  const float* b2   = (const float*)d_in[17];
  const float* g2w  = (const float*)d_in[18];
  const float* g2b  = (const float*)d_in[19];
  const float* g2ms = (const float*)d_in[20];
  const float* Wn   = (const float*)d_in[21];
  const float* bn   = (const float*)d_in[22];
  float* out = (float*)d_out;

  char* base = (char*)d_ws;
  size_t off = 0;
  auto take = [&](size_t bytes) { size_t p = off; off += (bytes + 255) & ~(size_t)255; return p; };

  // transposed f16 weights (persistent)
  size_t o_wqt = take((size_t)SDIM * HID * 2);          // 128x128
  size_t o_wkt = take((size_t)SEMDIM * HID * 2);        // Bt: 128x768
  size_t o_wvt = take((size_t)SEMDIM * HID * 2);
  size_t o_wot = take((size_t)256 * SDIM * 2);          // Bt: 128x256
  size_t o_w1t = take((size_t)INDIM * F1 * 2);          // Bt: 512x896
  size_t o_w2t = take((size_t)F1 * HID * 2);            // Bt: 128x512

  const size_t SZ_XH  = (size_t)NN * INDIM * 2;         // 89.6 MB
  const size_t SZ_H0  = (size_t)NN * SDIM * 4;          // 25.6 MB
  const size_t SZ_QKV = (size_t)3 * NN * HID * 4;       // 76.8 MB
  const size_t SZ_CAT = (size_t)NN * 256 * 2;           // 25.6 MB
  const size_t SZ_H2H = (size_t)NN * F1 * 2;            // 51.2 MB
  size_t o_R1 = take(SZ_XH + SZ_H0);                    // x_h+h0f -> out1 -> h3
  size_t o_R2 = take(SZ_QKV + SZ_CAT);                  // QKV+cat0 -> hw1 -> out2
  size_t o_R3 = take((size_t)NN * INDIM * 2);           // h_h -> h2_h + hw2

  size_t o_alpha1 = take((size_t)NT * NHEADS * 4);
  size_t o_amax1  = take((size_t)NN * NHEADS * 4);
  size_t o_den1   = take((size_t)NN * NHEADS * 4);
  size_t o_asrc1  = take((size_t)NN * NHEADS * 4);
  size_t o_adst1  = take((size_t)NN * NHEADS * 4);
  size_t o_alpha2 = take((size_t)NT * 4);
  size_t o_amax2  = take((size_t)NN * 4);
  size_t o_den2   = take((size_t)NN * 4);
  size_t o_asrc2  = take((size_t)NN * 4);
  size_t o_adst2  = take((size_t)NN * 4);
  size_t o_csum   = take((size_t)F1 * 4);
  size_t o_vsum   = take((size_t)F1 * 4);
  if (ws_size < off) return;

  _Float16* wqt = (_Float16*)(base + o_wqt);
  _Float16* wkt = (_Float16*)(base + o_wkt);
  _Float16* wvt = (_Float16*)(base + o_wvt);
  _Float16* wot = (_Float16*)(base + o_wot);
  _Float16* w1t = (_Float16*)(base + o_w1t);
  _Float16* w2t = (_Float16*)(base + o_w2t);

  _Float16* x_h  = (_Float16*)(base + o_R1);
  float*    h0f  = (float*)(base + o_R1 + SZ_XH);
  float*    out1 = (float*)(base + o_R1);
  float*    h3   = (float*)(base + o_R1);

  float*    Qf   = (float*)(base + o_R2);
  float*    Kf   = (float*)(base + o_R2 + (size_t)NN * HID * 4);
  float*    Vf   = (float*)(base + o_R2 + (size_t)2 * NN * HID * 4);
  _Float16* cat0 = (_Float16*)(base + o_R2 + SZ_QKV);
  float*    hw1  = (float*)(base + o_R2);
  float*    out2 = (float*)(base + o_R2);

  _Float16* h_h  = (_Float16*)(base + o_R3);
  _Float16* h2_h = (_Float16*)(base + o_R3);
  float*    hw2  = (float*)(base + o_R3 + SZ_H2H);

  float*    alpha1 = (float*)(base + o_alpha1);
  unsigned* amax1  = (unsigned*)(base + o_amax1);
  float*    den1   = (float*)(base + o_den1);
  float*    asrc1  = (float*)(base + o_asrc1);
  float*    adst1  = (float*)(base + o_adst1);
  float*    alpha2 = (float*)(base + o_alpha2);
  unsigned* amax2  = (unsigned*)(base + o_amax2);
  float*    den2   = (float*)(base + o_den2);
  float*    asrc2  = (float*)(base + o_asrc2);
  float*    adst2  = (float*)(base + o_adst2);
  float*    csum   = (float*)(base + o_csum);
  float*    vsum   = (float*)(base + o_vsum);

  auto gemm = [&](const _Float16* A, int lda, const _Float16* Bt, int ldb,
                  float* C, int ldc, int M, int K, int Nc) {
    int Mtiles = (M + 31) / 32, Ntiles = Nc / 32;
    int blocks = (Mtiles * Ntiles + 7) / 8;
    gemm_wmma_f16<<<blocks, 256, 0, stream>>>(A, lda, Bt, ldb, C, ldc, M, K, Ntiles);
  };

  const int GS = 4096;  // grid-stride blocks

  // --- phase 0: convert activations + weights ---
  k_f32_to_f16<<<GS, 256, 0, stream>>>(x, x_h, (long)NN * INDIM);
  k_transp<<<256, 256, 0, stream>>>(Wq, wqt, SDIM, HID);
  k_transp<<<512, 256, 0, stream>>>(Wk, wkt, SEMDIM, HID);
  k_transp<<<512, 256, 0, stream>>>(Wv, wvt, SEMDIM, HID);
  k_transp<<<256, 256, 0, stream>>>(Wo, wot, 256, SDIM);
  k_transp<<<1024, 256, 0, stream>>>(W1, w1t, INDIM, F1);
  k_transp<<<512, 256, 0, stream>>>(W2, w2t, F1, HID);

  // --- phase 1: cross-attention fusion ---
  gemm(x_h, INDIM, wqt, SDIM, Qf, HID, NN, SDIM, HID);          // Q = x_struct @ Wq
  gemm(x_h + SDIM, INDIM, wkt, SEMDIM, Kf, HID, NN, SEMDIM, HID);
  gemm(x_h + SDIM, INDIM, wvt, SEMDIM, Vf, HID, NN, SEMDIM, HID);
  k_fuse1<<<(NN + 7) / 8, 256, 0, stream>>>(x, Qf, Kf, Vf, cat0);
  gemm(cat0, 256, wot, 256, h0f, SDIM, NN, 256, SDIM);          // h0 = cat @ Wo
  k_build_h<<<GS, 256, 0, stream>>>(h0f, bo, x_h, h_h);

  // --- phase 2: GAT conv 1 (4 heads x 128) ---
  gemm(h_h, INDIM, w1t, INDIM, hw1, F1, NN, INDIM, F1);         // hw1 = h @ W1
  k_zero<<<GS, 256, 0, stream>>>(out1, (long)NN * F1);
  k_zero<<<256, 256, 0, stream>>>((float*)amax1, (long)NN * NHEADS);
  k_zero<<<256, 256, 0, stream>>>(den1, (long)NN * NHEADS);
  k_acoef<<<((long)NN * NHEADS + 255) / 256, 256, 0, stream>>>(hw1, as1, ad1, asrc1, adst1, NHEADS);
  k_edge_alpha<<<(NT + 255) / 256, 256, 0, stream>>>(ei, asrc1, adst1, alpha1, amax1, NHEADS);
  k_edge_exp<<<(NT + 255) / 256, 256, 0, stream>>>(ei, alpha1, amax1, den1, NHEADS);
  k_edge_scatter<<<(NT + 7) / 8, 256, 0, stream>>>(ei, hw1, alpha1, den1, out1, NHEADS);

  // graph_norm 1 + elu -> h2 (f16) ; also free R2 for out2
  k_zero<<<1, 256, 0, stream>>>(out2, 1);  // placeholder; real zero below
  k_zero<<<GS, 256, 0, stream>>>(out2, (long)NN * HID);
  k_zero<<<1, 256, 0, stream>>>(csum, F1);
  k_zero<<<1, 256, 0, stream>>>(vsum, F1);
  k_colsum<<<dim3((F1 + 255) / 256, 64), 256, 0, stream>>>(out1, b1, csum, F1);
  k_colvar<<<dim3((F1 + 255) / 256, 64), 256, 0, stream>>>(out1, b1, csum, g1ms, vsum, F1);
  k_norm<<<GS, 256, 0, stream>>>(out1, b1, csum, vsum, g1ms, g1w, g1b, (void*)h2_h, F1, 1);

  // --- phase 3: GAT conv 2 (1 head x 128) ---
  gemm(h2_h, F1, w2t, F1, hw2, HID, NN, F1, HID);               // hw2 = h2 @ W2
  k_zero<<<256, 256, 0, stream>>>((float*)amax2, NN);
  k_zero<<<256, 256, 0, stream>>>(den2, NN);
  k_acoef<<<(NN + 255) / 256, 256, 0, stream>>>(hw2, as2, ad2, asrc2, adst2, 1);
  k_edge_alpha<<<(NT + 255) / 256, 256, 0, stream>>>(ei, asrc2, adst2, alpha2, amax2, 1);
  k_edge_exp<<<(NT + 255) / 256, 256, 0, stream>>>(ei, alpha2, amax2, den2, 1);
  k_edge_scatter<<<(NT + 7) / 8, 256, 0, stream>>>(ei, hw2, alpha2, den2, out2, 1);

  // graph_norm 2 + elu -> h3 (f32)
  k_zero<<<1, 256, 0, stream>>>(csum, HID);
  k_zero<<<1, 256, 0, stream>>>(vsum, HID);
  k_colsum<<<dim3(1, 64), 256, 0, stream>>>(out2, b2, csum, HID);
  k_colvar<<<dim3(1, 64), 256, 0, stream>>>(out2, b2, csum, g2ms, vsum, HID);
  k_norm<<<GS, 256, 0, stream>>>(out2, b2, csum, vsum, g2ms, g2w, g2b, (void*)h3, HID, 0);

  // --- output projection ---
  k_final<<<(NN + 255) / 256, 256, 0, stream>>>(h3, Wn, bn, out);
}
